// GroupQueryAttention_4329327034815
// MI455X (gfx1250) — matrix operand order repaired and checked by reading, not compiled, not run on hardware
//
#include <hip/hip_runtime.h>
#include <hip/hip_bf16.h>

// ---------------- problem constants (match reference) ----------------
#define B_     2
#define S_     2048
#define DIN_   1024
#define HID_   1024
#define H_     16
#define HKV_   4
#define R_     4
#define HD_    64
#define SCALE_ 0.125f      // 64^-0.5
#define LOG2E_ 1.44269504088896340736f

typedef __attribute__((ext_vector_type(16))) __bf16 v16bf;
typedef __attribute__((ext_vector_type(8)))  float  v8f;
typedef __attribute__((ext_vector_type(4)))  unsigned int u32x4;
typedef __attribute__((ext_vector_type(4)))  int i32x4;
typedef __attribute__((ext_vector_type(8)))  int i32x8;

union FragU { v16bf v; unsigned int u[8]; __bf16 h[16]; };

#if __has_builtin(__builtin_amdgcn_tensor_load_to_lds) && \
    __has_builtin(__builtin_amdgcn_s_wait_tensorcnt)
#define HAVE_TDM 1
#else
#define HAVE_TDM 0
#endif

__device__ __forceinline__ v8f wmma_bf16(v16bf a, v16bf b, v8f c) {
  return __builtin_amdgcn_wmma_f32_16x16x32_bf16(
      /*neg_a=*/false, a, /*neg_b=*/false, b,
      /*c_mod=*/(short)0, c, /*reuse_a=*/false, /*reuse_b=*/false);
}

#if HAVE_TDM
__device__ __forceinline__ unsigned lds_offset(const void* p) {
  // flat LDS pointers carry the logical LDS byte offset in the low 32 bits
  return (unsigned)(unsigned long long)p;
}

// Issue a 2D TDM tile load: bf16 tensor [td1 rows][td0 elems], row stride
// stride0 (elements); tile [tile1 rows][tile0 elems] starting at gptr, packed
// row-major into LDS at lds_off. D# layout per CDNA5 ISA ch.8.
__device__ __forceinline__ void tdm_issue_tile(const void* gptr, unsigned lds_off,
                                               unsigned td0, unsigned td1,
                                               unsigned tile0, unsigned tile1,
                                               unsigned stride0) {
  unsigned long long ga = (unsigned long long)gptr;
  u32x4 g0;
  g0[0] = 1u;                                    // count=1, user descriptor
  g0[1] = lds_off;                               // lds_addr (bytes)
  g0[2] = (unsigned)(ga & 0xFFFFFFFFull);        // global_addr[31:0]
  g0[3] = (unsigned)((ga >> 32) & 0x1FFFFFFull)  // global_addr[56:32]
          | (2u << 30);                          // type = 2 ("image")
  i32x8 g1;
  g1[0] = 0x00010000;                            // workgroup_mask=0, data_size=2B
  g1[1] = (int)(td0 << 16);                      // tensor_dim0[15:0]
  g1[2] = (int)((td0 >> 16) | (td1 << 16));      // tensor_dim0[31:16], dim1[15:0]
  g1[3] = (int)((td1 >> 16) | (tile0 << 16));    // tensor_dim1[31:16], tile_dim0
  g1[4] = (int)tile1;                            // tile_dim1 (tile_dim2 = 0)
  g1[5] = (int)stride0;                          // tensor_dim0_stride[31:0]
  g1[6] = 0;                                     // stride0 hi, dim1_stride lo
  g1[7] = 0;                                     // dim1_stride hi
  i32x4 z4 = {0, 0, 0, 0};
#if __clang_major__ >= 23
  i32x8 z8 = {0, 0, 0, 0, 0, 0, 0, 0};
  __builtin_amdgcn_tensor_load_to_lds(g0, g1, z4, z4, z8, 0);
#else
  __builtin_amdgcn_tensor_load_to_lds(g0, g1, z4, z4, 0);
#endif
}
#endif // HAVE_TDM

// A fragment (16x32 bf16, M x K) from row-major memory, stride lda (elements).
__device__ __forceinline__ v16bf load_a_rowmajor(const __bf16* base, int lda) {
  int lane = threadIdx.x & 31;
  int row  = lane & 15;
  int half = lane >> 4;
  FragU f;
#pragma unroll
  for (int i = 0; i < 8; ++i) {
    int k = ((i & 4) ? 16 : 0) + half * 8 + (i & 3) * 2;
    f.u[i] = *(const unsigned int*)(base + row * lda + k);
  }
  return f.v;
}

// B fragment (32x16 bf16, K x N) from N-major memory (one N-row contiguous in K).
__device__ __forceinline__ v16bf load_b_nmajor(const __bf16* base, int rowstride) {
  int lane = threadIdx.x & 31;
  int col  = lane & 15;
  int kb   = (lane >> 4) * 8;
  FragU f;
#pragma unroll
  for (int i = 0; i < 8; ++i)
    f.u[i] = *(const unsigned int*)(base + col * rowstride + kb + ((i & 4) ? 16 : 0) + (i & 3) * 2);
  return f.v;
}

// max-reduction across the 16 lanes of a half-wave (butterfly).
#if __has_builtin(__builtin_amdgcn_permlane16)
__device__ __forceinline__ float permf16(float x, unsigned lo, unsigned hi) {
  unsigned xi = __builtin_bit_cast(unsigned, x);
  unsigned r  = __builtin_amdgcn_permlane16(xi, xi, lo, hi, false, false);
  return __builtin_bit_cast(float, r);
}
__device__ __forceinline__ float rowmax16(float x) {
  x = fmaxf(x, permf16(x, 0x67452301u, 0xEFCDAB89u)); // xor 1
  x = fmaxf(x, permf16(x, 0x54761032u, 0xDCFE98BAu)); // xor 2
  x = fmaxf(x, permf16(x, 0x32107654u, 0xBA98FEDCu)); // xor 4
  x = fmaxf(x, permf16(x, 0xFEDCBA98u, 0x76543210u)); // xor 8
  return x;
}
#else
__device__ __forceinline__ float rowmax16(float x) {
#pragma unroll
  for (int off = 1; off < 16; off <<= 1) x = fmaxf(x, __shfl_xor(x, off, 32));
  return x;
}
#endif

// ---------------- f32 -> bf16 cast (x8 vectorized, optional scale) ----------------
__global__ __launch_bounds__(256) void cast_f32_to_bf16(const float* __restrict__ in,
                                                        __bf16* __restrict__ out,
                                                        int n, float scale) {
  int i = (blockIdx.x * 256 + threadIdx.x) * 8;
  if (i >= n) return;
  float4 a = *(const float4*)&in[i];
  float4 b = *(const float4*)&in[i + 4];
  union { uint4 q; __bf16 h[8]; } u;
  u.h[0] = (__bf16)(a.x * scale); u.h[1] = (__bf16)(a.y * scale);
  u.h[2] = (__bf16)(a.z * scale); u.h[3] = (__bf16)(a.w * scale);
  u.h[4] = (__bf16)(b.x * scale); u.h[5] = (__bf16)(b.y * scale);
  u.h[6] = (__bf16)(b.z * scale); u.h[7] = (__bf16)(b.w * scale);
  *(uint4*)&out[i] = u.q;
}

// ---- f32 [K][N] -> bf16 [N][K] transposed cast (one-time, weights only) ----
__global__ __launch_bounds__(256) void cast_transpose_bf16(const float* __restrict__ in,
                                                           __bf16* __restrict__ out,
                                                           int K, int N, float scale) {
  int i = (blockIdx.x * 256 + threadIdx.x) * 8;
  if (i >= K * N) return;
  int k = i / N, n = i % N;     // 8 consecutive along N
  float4 a = *(const float4*)&in[i];
  float4 b = *(const float4*)&in[i + 4];
  out[(size_t)(n + 0) * K + k] = (__bf16)(a.x * scale);
  out[(size_t)(n + 1) * K + k] = (__bf16)(a.y * scale);
  out[(size_t)(n + 2) * K + k] = (__bf16)(a.z * scale);
  out[(size_t)(n + 3) * K + k] = (__bf16)(a.w * scale);
  out[(size_t)(n + 4) * K + k] = (__bf16)(b.x * scale);
  out[(size_t)(n + 5) * K + k] = (__bf16)(b.y * scale);
  out[(size_t)(n + 6) * K + k] = (__bf16)(b.z * scale);
  out[(size_t)(n + 7) * K + k] = (__bf16)(b.w * scale);
}

// ---------------- bf16 GEMM: C[M,N] = A[M,K] @ Wt[N,K]^T ----------------
// block tile 128x128, K-step 64; 8 waves as 4(M) x 2(N); wave tile 32x64.
// Tiles staged via TDM (double-buffered, overlapped with WMMA) when available.
// MODE: 0 = bf16 row-major out, 1 = f32 row-major out, 2 = bf16 transposed out.
template <int MODE>
__global__ __launch_bounds__(256) void gemm_bf16(const __bf16* __restrict__ A,
                                                 const __bf16* __restrict__ Wt,
                                                 void* __restrict__ Cout,
                                                 int M, int N, int K) {
  __shared__ __align__(16) __bf16 sA[2][128 * 64];    // [m][k]
  __shared__ __align__(16) __bf16 sWt[2][128 * 64];   // [n][k]

  const int tid = threadIdx.x;
  const int wid = tid >> 5;
  const int mw  = wid >> 1;          // 0..3
  const int nw  = wid & 1;           // 0..1
  const int m0  = blockIdx.x * 128;
  const int n0  = blockIdx.y * 128;

  v8f acc[2][4];
#pragma unroll
  for (int s = 0; s < 2; ++s)
#pragma unroll
    for (int nt = 0; nt < 4; ++nt) acc[s][nt] = v8f{};

  const int NKT = K >> 6;
#if HAVE_TDM
  if (wid == 0) {
    tdm_issue_tile(&A[(size_t)m0 * K], lds_offset(&sA[0][0]), K, M, 64, 128, K);
    tdm_issue_tile(&Wt[(size_t)n0 * K], lds_offset(&sWt[0][0]), K, N, 64, 128, K);
  }
#endif
  for (int t = 0; t < NKT; ++t) {
    const int k0  = t << 6;
    const int buf = t & 1;
#if HAVE_TDM
    if (wid == 0) {
      if (t + 1 < NKT) {
        tdm_issue_tile(&A[(size_t)m0 * K + k0 + 64],
                       lds_offset(&sA[buf ^ 1][0]), K, M, 64, 128, K);
        tdm_issue_tile(&Wt[(size_t)n0 * K + k0 + 64],
                       lds_offset(&sWt[buf ^ 1][0]), K, N, 64, 128, K);
        __builtin_amdgcn_s_wait_tensorcnt(2);   // current tile's 2 loads done
      } else {
        __builtin_amdgcn_s_wait_tensorcnt(0);
      }
    }
#else
#pragma unroll
    for (int i = tid * 8; i < 128 * 64; i += 2048) {
      int r = i >> 6, c = i & 63;
      *(uint4*)&sA[buf][i]  = *(const uint4*)&A[(size_t)(m0 + r) * K + (k0 + c)];
      *(uint4*)&sWt[buf][i] = *(const uint4*)&Wt[(size_t)(n0 + r) * K + (k0 + c)];
    }
#endif
    __syncthreads();

    v16bf a[2][2];
#pragma unroll
    for (int s = 0; s < 2; ++s)
#pragma unroll
      for (int kh = 0; kh < 2; ++kh)
        a[s][kh] = load_a_rowmajor(&sA[buf][(mw * 32 + s * 16) * 64 + kh * 32], 64);
#pragma unroll
    for (int nt = 0; nt < 4; ++nt) {
      v16bf b0 = load_b_nmajor(&sWt[buf][(nw * 64 + nt * 16) * 64], 64);
      v16bf b1 = load_b_nmajor(&sWt[buf][(nw * 64 + nt * 16) * 64 + 32], 64);
      acc[0][nt] = wmma_bf16(a[0][0], b0, acc[0][nt]);
      acc[0][nt] = wmma_bf16(a[0][1], b1, acc[0][nt]);
      acc[1][nt] = wmma_bf16(a[1][0], b0, acc[1][nt]);
      acc[1][nt] = wmma_bf16(a[1][1], b1, acc[1][nt]);
    }
    __syncthreads();
  }

  const int lane = tid & 31;
  const int col  = lane & 15;
  const int half = lane >> 4;
#pragma unroll
  for (int s = 0; s < 2; ++s) {
#pragma unroll
    for (int nt = 0; nt < 4; ++nt) {
#pragma unroll
      for (int e = 0; e < 8; ++e) {
        size_t row = (size_t)(m0 + mw * 32 + s * 16 + e + 8 * half);
        size_t cc  = (size_t)(n0 + nw * 64 + nt * 16 + col);
        if (MODE == 1)      ((float*)Cout)[row * N + cc]  = acc[s][nt][e];
        else if (MODE == 2) ((__bf16*)Cout)[cc * M + row] = (__bf16)acc[s][nt][e];
        else                ((__bf16*)Cout)[row * N + cc] = (__bf16)acc[s][nt][e];
      }
    }
  }
}

// ---------------- flash attention (GQA) ----------------
// grid.x = b*16 + kv*4 + r, grid.y = S/128; 8 waves, 16 query rows/wave.
// 64-key tiles: 8 QK + 8 PV + 2 row-sum WMMAs per tile; K/V tiles staged by
// TDM (double-buffered). Scores pre-scaled by SCALE*log2e -> pure exp2 softmax.
__global__ __launch_bounds__(256) void attn_kernel(const __bf16* __restrict__ Q,
                                                   const __bf16* __restrict__ Km,
                                                   const __bf16* __restrict__ VT,
                                                   __bf16* __restrict__ O) {
  __shared__ __align__(16) __bf16 sK[2][64 * 64];    // [key][dim]
  __shared__ __align__(16) __bf16 sVt[2][64 * 64];   // [dim][key]
  __shared__ __align__(16) __bf16 sP[8][16 * 64];    // per-wave P [row][key]

  const int tid  = threadIdx.x;
  const int wid  = tid >> 5;
  const int lane = tid & 31;
  const int col  = lane & 15;
  const int half = lane >> 4;

  const int hrb  = blockIdx.x;
  const int b    = hrb >> 4;
  const int kv   = (hrb >> 2) & 3;
  const int head = hrb & 15;          // kv*R + r

  const int q0 = blockIdx.y * 128 + wid * 16;

  const __bf16* qbase = Q + (size_t)(b * S_ + q0) * HID_ + head * HD_;
  v16bf qa0 = load_a_rowmajor(qbase, HID_);        // dims 0..31
  v16bf qa1 = load_a_rowmajor(qbase + 32, HID_);   // dims 32..63

  FragU onesu;
#pragma unroll
  for (int i = 0; i < 16; ++i) onesu.h[i] = (__bf16)1.0f;
  const v16bf ones = onesu.v;

  v8f o[4] = {v8f{}, v8f{}, v8f{}, v8f{}};
  v8f lacc = v8f{};
  float mrow[8];
#pragma unroll
  for (int e = 0; e < 8; ++e) mrow[e] = -3.0e38f;

  const int NT = S_ >> 6;
#if HAVE_TDM
  if (wid == 0) {
    tdm_issue_tile(&Km[(size_t)(b * S_) * (HKV_ * HD_) + kv * HD_],
                   lds_offset(&sK[0][0]), HKV_ * HD_, B_ * S_, 64, 64, HKV_ * HD_);
    tdm_issue_tile(&VT[(size_t)(kv * HD_) * (B_ * S_) + b * S_],
                   lds_offset(&sVt[0][0]), B_ * S_, HKV_ * HD_, 64, 64, B_ * S_);
  }
#endif
  for (int t = 0; t < NT; ++t) {
    const int k0  = t << 6;
    const int buf = t & 1;
#if HAVE_TDM
    if (wid == 0) {
      if (t + 1 < NT) {
        tdm_issue_tile(&Km[(size_t)(b * S_ + k0 + 64) * (HKV_ * HD_) + kv * HD_],
                       lds_offset(&sK[buf ^ 1][0]),
                       HKV_ * HD_, B_ * S_, 64, 64, HKV_ * HD_);
        tdm_issue_tile(&VT[(size_t)(kv * HD_) * (B_ * S_) + b * S_ + k0 + 64],
                       lds_offset(&sVt[buf ^ 1][0]),
                       B_ * S_, HKV_ * HD_, 64, 64, B_ * S_);
        __builtin_amdgcn_s_wait_tensorcnt(2);
      } else {
        __builtin_amdgcn_s_wait_tensorcnt(0);
      }
    }
#else
#pragma unroll
    for (int i = tid * 8; i < 64 * 64; i += 2048) {
      int r = i >> 6, c = i & 63;
      *(uint4*)&sK[buf][i] = *(const uint4*)
          &Km[(size_t)(b * S_ + k0 + r) * (HKV_ * HD_) + kv * HD_ + c];
      *(uint4*)&sVt[buf][i] = *(const uint4*)
          &VT[(size_t)(kv * HD_ + r) * (B_ * S_) + b * S_ + k0 + c];
    }
#endif
    __syncthreads();

    // S = Q K^T : score frag j, column c <-> key 4c+j
    v8f s[4];
#pragma unroll
    for (int j = 0; j < 4; ++j) {
      v16bf bk0 = load_b_nmajor(&sK[buf][j * 64], 256);        // dims 0..31
      v16bf bk1 = load_b_nmajor(&sK[buf][j * 64 + 32], 256);   // dims 32..63
      s[j] = wmma_bf16(qa0, bk0, v8f{});
      s[j] = wmma_bf16(qa1, bk1, s[j]);
    }

    // online softmax; only the row-max needs cross-lane work (permlane16)
#pragma unroll
    for (int e = 0; e < 8; ++e) {
      float t0 = s[0][e], t1 = s[1][e], t2 = s[2][e], t3 = s[3][e];
      float mx = rowmax16(fmaxf(fmaxf(t0, t1), fmaxf(t2, t3)));
      float mnew  = fmaxf(mrow[e], mx);
      float alpha = __builtin_amdgcn_exp2f(mrow[e] - mnew);
      mrow[e] = mnew;
      float p0 = __builtin_amdgcn_exp2f(t0 - mnew);
      float p1 = __builtin_amdgcn_exp2f(t1 - mnew);
      float p2 = __builtin_amdgcn_exp2f(t2 - mnew);
      float p3 = __builtin_amdgcn_exp2f(t3 - mnew);
      o[0][e] *= alpha; o[1][e] *= alpha; o[2][e] *= alpha; o[3][e] *= alpha;
      lacc[e] *= alpha;
      union { __bf16 h[4]; unsigned long long q; } pk;
      pk.h[0] = (__bf16)p0; pk.h[1] = (__bf16)p1;
      pk.h[2] = (__bf16)p2; pk.h[3] = (__bf16)p3;
      *(unsigned long long*)&sP[wid][(e + 8 * half) * 64 + 4 * col] = pk.q;
    }

    // O += P @ V ; row-sum l += P @ ones (matrix-pipe instead of shuffles)
    v16bf pa0 = load_a_rowmajor(&sP[wid][0], 64);    // keys 0..31
    v16bf pa1 = load_a_rowmajor(&sP[wid][32], 64);   // keys 32..63
#pragma unroll
    for (int nt = 0; nt < 4; ++nt) {
      v16bf bv0 = load_b_nmajor(&sVt[buf][(nt * 16) * 64], 64);        // keys 0..31
      v16bf bv1 = load_b_nmajor(&sVt[buf][(nt * 16) * 64 + 32], 64);   // keys 32..63
      o[nt] = wmma_bf16(pa0, bv0, o[nt]);
      o[nt] = wmma_bf16(pa1, bv1, o[nt]);
    }
    lacc = wmma_bf16(pa0, ones, lacc);
    lacc = wmma_bf16(pa1, ones, lacc);
    __syncthreads();
  }

  // normalize and write O (bf16, [B,S,HID] with head offset)
#pragma unroll
  for (int e = 0; e < 8; ++e) {
    float inv = 1.0f / lacc[e];
    int row   = q0 + e + 8 * half;
    size_t base = (size_t)(b * S_ + row) * HID_ + head * HD_ + col;
    O[base + 0]  = (__bf16)(o[0][e] * inv);
    O[base + 16] = (__bf16)(o[1][e] * inv);
    O[base + 32] = (__bf16)(o[2][e] * inv);
    O[base + 48] = (__bf16)(o[3][e] * inv);
  }
}

// ---------------- host driver ----------------
extern "C" void kernel_launch(void* const* d_in, const int* in_sizes, int n_in,
                              void* d_out, int out_size, void* d_ws, size_t ws_size,
                              hipStream_t stream) {
  const float* x  = (const float*)d_in[0];
  const float* Wq = (const float*)d_in[1];
  const float* Wk = (const float*)d_in[2];
  const float* Wv = (const float*)d_in[3];
  const float* Wo = (const float*)d_in[4];

  const size_t M   = (size_t)B_ * S_;              // 4096
  const size_t nX  = M * DIN_;                     // 4,194,304
  const size_t nWq = (size_t)DIN_ * HID_;          // 1,048,576
  const size_t nWk = (size_t)DIN_ * (HKV_ * HD_);  // 262,144
  const size_t nWo = (size_t)HID_ * DIN_;          // 1,048,576
  const size_t nQ  = M * HID_;
  const size_t nK  = M * (HKV_ * HD_);

  __bf16* w   = (__bf16*)d_ws;
  __bf16* xb  = w;  w += nX;
  __bf16* wqt = w;  w += nWq;   // [HID][DIN]
  __bf16* wkt = w;  w += nWk;   // [HKV*HD][DIN]
  __bf16* wvt = w;  w += nWk;
  __bf16* wot = w;  w += nWo;   // [DOUT][HID]
  __bf16* qb  = w;  w += nQ;    // [B*S][HID]
  __bf16* kb  = w;  w += nK;    // [B*S][HKV*HD]
  __bf16* vt  = w;  w += nK;    // [HKV*HD][B*S]  (transposed V)
  __bf16* ob  = w;  w += nQ;    // [B*S][HID]

  // 1) casts; weights transposed to N-major; fold SCALE*log2e into Wq
  cast_f32_to_bf16<<<(int)(nX / 2048), 256, 0, stream>>>(x, xb, (int)nX, 1.0f);
  cast_transpose_bf16<<<(int)(nWq / 2048), 256, 0, stream>>>(Wq, wqt, DIN_, HID_,
                                                             SCALE_ * LOG2E_);
  cast_transpose_bf16<<<(int)(nWk / 2048), 256, 0, stream>>>(Wk, wkt, DIN_,
                                                             HKV_ * HD_, 1.0f);
  cast_transpose_bf16<<<(int)(nWk / 2048), 256, 0, stream>>>(Wv, wvt, DIN_,
                                                             HKV_ * HD_, 1.0f);
  cast_transpose_bf16<<<(int)(nWo / 2048), 256, 0, stream>>>(Wo, wot, HID_, DIN_,
                                                             1.0f);

  // 2) QKV projections (V written transposed)
  gemm_bf16<0><<<dim3(M / 128, HID_ / 128), 256, 0, stream>>>(
      xb, wqt, (void*)qb, (int)M, HID_, DIN_);
  gemm_bf16<0><<<dim3(M / 128, (HKV_ * HD_) / 128), 256, 0, stream>>>(
      xb, wkt, (void*)kb, (int)M, HKV_ * HD_, DIN_);
  gemm_bf16<2><<<dim3(M / 128, (HKV_ * HD_) / 128), 256, 0, stream>>>(
      xb, wvt, (void*)vt, (int)M, HKV_ * HD_, DIN_);

  // 3) attention
  attn_kernel<<<dim3(B_ * H_, S_ / 128), 256, 0, stream>>>(qb, kb, vt, ob);

  // 4) output projection -> f32 d_out
  gemm_bf16<1><<<dim3(M / 128, DIN_ / 128), 256, 0, stream>>>(
      ob, wot, d_out, (int)M, DIN_, HID_);
}
